// MinLSTM_27487790695070
// MI455X (gfx1250) — matrix operand order repaired and checked by reading, not compiled, not run on hardware
//
#include <hip/hip_runtime.h>
#include <math.h>

typedef __attribute__((ext_vector_type(16))) _Float16 v16h;
typedef __attribute__((ext_vector_type(8)))  _Float16 v8h;
typedef __attribute__((ext_vector_type(8)))  float    v8f;
typedef __attribute__((ext_vector_type(4)))  float    v4f;

// Problem constants
constexpr int BATCH = 16;
constexpr int SEQ   = 2048;
constexpr int IN    = 256;
constexpr int HID   = 512;
constexpr int K0    = IN + HID;   // 768  (layer0 K)
constexpr int K1    = HID + HID;  // 1024 (layer1 K)
constexpr int K0P   = K0 + 8;     // padded LDS stride (halves)
constexpr int K1P   = K1 + 8;
constexpr int NSLICE = 32;        // WGs per layer (16 h-cols each)
constexpr int NWG    = 64;        // total persistent WGs

static __device__ __forceinline__ v16h pack16(v8h r0, v8h r1) {
  v16h a;
#pragma unroll
  for (int i = 0; i < 8; ++i) { a[i] = r0[i]; a[8 + i] = r1[i]; }
  return a;
}

static __device__ __forceinline__ v8h cvt_f32x8_to_h(const float* __restrict__ p) {
  v4f a = *(const v4f*)p;
  v4f b = *(const v4f*)(p + 4);
  v8h r;
#pragma unroll
  for (int i = 0; i < 4; ++i) { r[i] = (_Float16)a[i]; r[4 + i] = (_Float16)b[i]; }
  return r;
}

static __device__ __forceinline__ float sigmoidf_(float x) {
  return 1.0f / (1.0f + __expf(-x));
}

__global__ void __launch_bounds__(128) lstm_init_kernel(_Float16* h0buf, _Float16* h1buf,
                                                        unsigned* barCnt, unsigned* barGen) {
  const int tid = blockIdx.x * blockDim.x + threadIdx.x;
  const int total = 2 * BATCH * HID;  // per buffer (double-buffered)
  for (int i = tid; i < total; i += gridDim.x * blockDim.x) {
    h0buf[i] = (_Float16)0.0f;
    h1buf[i] = (_Float16)0.0f;
  }
  if (tid == 0) { *barCnt = 0u; *barGen = 0u; }
}

__global__ void __launch_bounds__(128) lstm_persistent_kernel(
    const float* __restrict__ x,
    const float* __restrict__ W0, const float* __restrict__ b0,
    const float* __restrict__ W1, const float* __restrict__ b1,
    float* __restrict__ out,
    _Float16* __restrict__ h0buf,   // [2][BATCH*HID] f16, ping-pong
    _Float16* __restrict__ h1buf,   // [2][BATCH*HID] f16, ping-pong
    unsigned* __restrict__ barCnt, unsigned* __restrict__ barGen) {
  // 64 cols * (K1+8) halves = 132 KB; + gates 4 KB + c-state 1 KB (<320 KB WGP LDS)
  __shared__ _Float16 wlds[64 * K1P];
  __shared__ float gatesS[4][256];   // [gate][m*16+n] fp32 tiles
  __shared__ float cS[256];          // this WG's c slice [b*16+j]
  __shared__ float biasS[4][16];

  const int tid   = threadIdx.x;
  const int lane  = tid & 31;
  const int wv    = tid >> 5;                 // wave -> gate tile (i,f,c~,o)
  const bool isL0 = (blockIdx.x < NSLICE);
  const int slice = isL0 ? (int)blockIdx.x : (int)blockIdx.x - NSLICE;
  const int K     = isL0 ? K0 : K1;
  const int Kpad  = isL0 ? K0P : K1P;
  const float* W    = isL0 ? W0 : W1;
  const float* bias = isL0 ? b0 : b1;

  // ---- one-time: load this WG's weight slice into LDS (fp32 -> f16, K-major) ----
  for (int c = 0; c < 64; ++c) {
    const int grow = (c >> 4) * HID + slice * 16 + (c & 15);
    const float* __restrict__ wrow = W + (size_t)grow * K;
    _Float16* __restrict__ dst = wlds + (size_t)c * Kpad;
    for (int k = tid; k < K; k += 128) dst[k] = (_Float16)wrow[k];
  }
  if (tid < 64) biasS[tid >> 4][tid & 15] = bias[(tid >> 4) * HID + slice * 16 + (tid & 15)];
  for (int i = tid; i < 256; i += 128) cS[i] = 0.0f;
  __syncthreads();

  const int n  = lane & 15;   // B-frag column / C-frag column
  const int hi = lane >> 4;
  const int m  = lane & 15;   // A-frag row (batch)
  const size_t outTail = (size_t)BATCH * SEQ * HID;
  const _Float16* __restrict__ wl = wlds + (size_t)(wv * 16 + n) * Kpad + hi * 16;

  for (int p = 0; p <= SEQ; ++p) {
    const bool active = isL0 ? (p < SEQ) : (p >= 1);
    if (active) {
      // Two independent accumulators -> 2-way ILP in the XDL pipe
      v8f acc0 = {};
      v8f acc1 = {};
      if (isL0) {
        // --- layer0: A = [x_t | h0^(p)] ---
        const float* __restrict__ xrow = x + ((size_t)m * SEQ + p) * IN;
#pragma unroll
        for (int kb = 0; kb < IN; kb += 64) {
          v8h a0 = cvt_f32x8_to_h(xrow + kb + hi * 16);
          v8h a1 = cvt_f32x8_to_h(xrow + kb + 8 + hi * 16);
          v8h b0v = *(const v8h*)(wl + kb);
          v8h b1v = *(const v8h*)(wl + kb + 8);
          acc0 = __builtin_amdgcn_wmma_f32_16x16x32_f16(false, pack16(a0, a1),
                                                        false, pack16(b0v, b1v),
                                                        (short)0, acc0, false, false);
          v8h a2 = cvt_f32x8_to_h(xrow + kb + 32 + hi * 16);
          v8h a3 = cvt_f32x8_to_h(xrow + kb + 40 + hi * 16);
          v8h b2v = *(const v8h*)(wl + kb + 32);
          v8h b3v = *(const v8h*)(wl + kb + 40);
          acc1 = __builtin_amdgcn_wmma_f32_16x16x32_f16(false, pack16(a2, a3),
                                                        false, pack16(b2v, b3v),
                                                        (short)0, acc1, false, false);
        }
        const _Float16* __restrict__ hrow =
            h0buf + (size_t)(p & 1) * (BATCH * HID) + (size_t)m * HID;
#pragma unroll
        for (int kb = 0; kb < HID; kb += 64) {
          v8h a0 = *(const v8h*)(hrow + kb + hi * 16);
          v8h a1 = *(const v8h*)(hrow + kb + 8 + hi * 16);
          v8h b0v = *(const v8h*)(wl + IN + kb);
          v8h b1v = *(const v8h*)(wl + IN + kb + 8);
          acc0 = __builtin_amdgcn_wmma_f32_16x16x32_f16(false, pack16(a0, a1),
                                                        false, pack16(b0v, b1v),
                                                        (short)0, acc0, false, false);
          v8h a2 = *(const v8h*)(hrow + kb + 32 + hi * 16);
          v8h a3 = *(const v8h*)(hrow + kb + 40 + hi * 16);
          v8h b2v = *(const v8h*)(wl + IN + kb + 32);
          v8h b3v = *(const v8h*)(wl + IN + kb + 40);
          acc1 = __builtin_amdgcn_wmma_f32_16x16x32_f16(false, pack16(a2, a3),
                                                        false, pack16(b2v, b3v),
                                                        (short)0, acc1, false, false);
        }
      } else {
        // --- layer1: A = [h0^(p) | h1^(p-1)] ---
        const _Float16* __restrict__ h0row =
            h0buf + (size_t)(p & 1) * (BATCH * HID) + (size_t)m * HID;
#pragma unroll
        for (int kb = 0; kb < HID; kb += 64) {
          v8h a0 = *(const v8h*)(h0row + kb + hi * 16);
          v8h a1 = *(const v8h*)(h0row + kb + 8 + hi * 16);
          v8h b0v = *(const v8h*)(wl + kb);
          v8h b1v = *(const v8h*)(wl + kb + 8);
          acc0 = __builtin_amdgcn_wmma_f32_16x16x32_f16(false, pack16(a0, a1),
                                                        false, pack16(b0v, b1v),
                                                        (short)0, acc0, false, false);
          v8h a2 = *(const v8h*)(h0row + kb + 32 + hi * 16);
          v8h a3 = *(const v8h*)(h0row + kb + 40 + hi * 16);
          v8h b2v = *(const v8h*)(wl + kb + 32);
          v8h b3v = *(const v8h*)(wl + kb + 40);
          acc1 = __builtin_amdgcn_wmma_f32_16x16x32_f16(false, pack16(a2, a3),
                                                        false, pack16(b2v, b3v),
                                                        (short)0, acc1, false, false);
        }
        const _Float16* __restrict__ h1row =
            h1buf + (size_t)((p + 1) & 1) * (BATCH * HID) + (size_t)m * HID;
#pragma unroll
        for (int kb = 0; kb < HID; kb += 64) {
          v8h a0 = *(const v8h*)(h1row + kb + hi * 16);
          v8h a1 = *(const v8h*)(h1row + kb + 8 + hi * 16);
          v8h b0v = *(const v8h*)(wl + HID + kb);
          v8h b1v = *(const v8h*)(wl + HID + kb + 8);
          acc0 = __builtin_amdgcn_wmma_f32_16x16x32_f16(false, pack16(a0, a1),
                                                        false, pack16(b0v, b1v),
                                                        (short)0, acc0, false, false);
          v8h a2 = *(const v8h*)(h1row + kb + 32 + hi * 16);
          v8h a3 = *(const v8h*)(h1row + kb + 40 + hi * 16);
          v8h b2v = *(const v8h*)(wl + HID + kb + 32);
          v8h b3v = *(const v8h*)(wl + HID + kb + 40);
          acc1 = __builtin_amdgcn_wmma_f32_16x16x32_f16(false, pack16(a2, a3),
                                                        false, pack16(b2v, b3v),
                                                        (short)0, acc1, false, false);
        }
      }
      // C/D layout: VGPR r holds (M = r + 8*hi, N = n)
#pragma unroll
      for (int r = 0; r < 8; ++r) gatesS[wv][(r + hi * 8) * 16 + n] = acc0[r] + acc1[r];
    }
    __syncthreads();

    if (active) {
      // elementwise LSTM cell: 256 elements, 2 per thread
#pragma unroll
      for (int e = 0; e < 2; ++e) {
        const int idx = tid + e * 128;
        const int bb = idx >> 4, j = idx & 15;
        const int col = slice * 16 + j;
        const float ig = gatesS[0][idx] + biasS[0][j];
        const float fg = gatesS[1][idx] + biasS[1][j];
        const float cg = gatesS[2][idx] + biasS[2][j];
        const float og = gatesS[3][idx] + biasS[3][j];
        const float cn = sigmoidf_(fg) * cS[idx] + sigmoidf_(ig) * tanhf(cg);
        const float hn = sigmoidf_(og) * tanhf(cn);
        cS[idx] = cn;
        if (isL0) {
          h0buf[(size_t)((p + 1) & 1) * (BATCH * HID) + (size_t)bb * HID + col] = (_Float16)hn;
          if (p == SEQ - 1) {  // final h0^(S), c0^(S)
            out[outTail + (size_t)bb * HID + col] = hn;
            out[outTail + (size_t)(2 * BATCH * HID) + (size_t)bb * HID + col] = cn;
          }
        } else {
          h1buf[(size_t)(p & 1) * (BATCH * HID) + (size_t)bb * HID + col] = (_Float16)hn;
          out[((size_t)bb * SEQ + (p - 1)) * HID + col] = hn;  // outputs[b, t, h]
          if (p == SEQ) {      // final h1^(S), c1^(S)
            out[outTail + (size_t)(BATCH * HID) + (size_t)bb * HID + col] = hn;
            out[outTail + (size_t)(3 * BATCH * HID) + (size_t)bb * HID + col] = cn;
          }
        }
      }
    }

    // ---- grid barrier (sense via monotonically increasing generation) ----
    __syncthreads();
    if (tid == 0) {
      __threadfence();
      const unsigned arrived = atomicAdd(barCnt, 1u) + 1u;
      if (arrived == (unsigned)NWG) {
        __hip_atomic_store(barCnt, 0u, __ATOMIC_RELAXED, __HIP_MEMORY_SCOPE_AGENT);
        __threadfence();
        atomicAdd(barGen, 1u);
      } else {
        while (__hip_atomic_load(barGen, __ATOMIC_ACQUIRE, __HIP_MEMORY_SCOPE_AGENT) <=
               (unsigned)p) {
          __builtin_amdgcn_s_sleep(2);
        }
      }
      __threadfence();
    }
    __syncthreads();
  }
}

extern "C" void kernel_launch(void* const* d_in, const int* in_sizes, int n_in,
                              void* d_out, int out_size, void* d_ws, size_t ws_size,
                              hipStream_t stream) {
  (void)in_sizes; (void)n_in; (void)out_size; (void)ws_size;
  const float* x  = (const float*)d_in[0];
  const float* W0 = (const float*)d_in[1];
  const float* b0 = (const float*)d_in[2];
  const float* W1 = (const float*)d_in[3];
  const float* b1 = (const float*)d_in[4];
  float* out = (float*)d_out;

  char* ws = (char*)d_ws;
  _Float16* h0buf = (_Float16*)ws;                               // 32768 B
  _Float16* h1buf = (_Float16*)(ws + 2 * BATCH * HID * 2);       // 32768 B
  unsigned* barCnt = (unsigned*)(ws + 2 * 2 * BATCH * HID * 2);  // 8 B
  unsigned* barGen = barCnt + 1;

  hipLaunchKernelGGL(lstm_init_kernel, dim3(1), dim3(128), 0, stream,
                     h0buf, h1buf, barCnt, barGen);
  hipLaunchKernelGGL(lstm_persistent_kernel, dim3(NWG), dim3(128), 0, stream,
                     x, W0, b0, W1, b1, out, h0buf, h1buf, barCnt, barGen);
}
